// LSTMModel_25262997635531
// MI455X (gfx1250) — matrix operand order repaired and checked by reading, compiled, not run on hardware
//
#include <hip/hip_runtime.h>
#include <hip/hip_bf16.h>

// ---------------------------------------------------------------------------
// BiLSTM tagger for gfx1250 (MI455X), wave32 + WMMA f16 16x16x32.
//
// Dims: V=32000 E=300 H=300 O=9 B=128 T=512.
// Padded: Hp=304 (16-mult), Np=4*Hp=1216, K padded to 320 (32-mult).
// Zero padding is exact through the LSTM: zero weights/bias => gates 0 =>
// c stays 0, h = 0.5*tanh(0) = 0 in padded columns.
// ---------------------------------------------------------------------------

typedef __attribute__((ext_vector_type(16))) _Float16 v16h;
typedef __attribute__((ext_vector_type(8)))  _Float16 v8h;
typedef __attribute__((ext_vector_type(8)))  float    v8f;

#define B_   128
#define T_   512
#define E_   300
#define H_   300
#define O_   9
#define EP   320          // padded K for emb / W_ih  (also KP for W_hh)
#define KP   320
#define HP   304          // padded hidden
#define NP   1216         // 4*HP gate columns
#define BT   65536        // B_*T_
#define HC   608          // 2*HP concat width
#define NGROUPS 152       // 8 m-tiles * 19 n-tiles (per-gate)

__device__ __forceinline__ float sig_(float x) { return 1.f / (1.f + __expf(-x)); }
__device__ __forceinline__ float th_(float x)  { return 2.f / (1.f + __expf(-2.f * x)) - 1.f; }

// ---------------------------------------------------------------------------
// Weight prep: f32 -> f16, pad H 300->304 and K 300->320 with zeros, K-major.
// ---------------------------------------------------------------------------
__global__ void prep_weights(const float* __restrict__ Wih_f, const float* __restrict__ Whh_f,
                             const float* __restrict__ b_f,
                             const float* __restrict__ Wih_b, const float* __restrict__ Whh_b,
                             const float* __restrict__ b_b,
                             const float* __restrict__ Wlin,
                             _Float16* __restrict__ Wihp, _Float16* __restrict__ Whhp,
                             float* __restrict__ biasp, float* __restrict__ Wlinp)
{
    int tid = blockIdx.x * blockDim.x + threadIdx.x;
    int nth = gridDim.x * blockDim.x;
    // Wihp / Whhp : [2][NP][EP]  (EP == KP)
    for (int i = tid; i < 2 * NP * EP; i += nth) {
        int d   = i / (NP * EP);
        int rem = i % (NP * EP);
        int r = rem / EP, k = rem % EP;
        int q = r / HP, j = r % HP;
        const float* Wih = d ? Wih_b : Wih_f;
        const float* Whh = d ? Whh_b : Whh_f;
        float vi = (j < H_ && k < E_) ? Wih[(q * H_ + j) * E_ + k] : 0.f;
        float vh = (j < H_ && k < H_) ? Whh[(q * H_ + j) * H_ + k] : 0.f;
        Wihp[i] = (_Float16)vi;
        Whhp[i] = (_Float16)vh;
    }
    // biasp : [2][NP]
    for (int i = tid; i < 2 * NP; i += nth) {
        int d = i / NP, r = i % NP;
        int q = r / HP, j = r % HP;
        const float* bb = d ? b_b : b_f;
        biasp[i] = (j < H_) ? bb[q * H_ + j] : 0.f;
    }
    // Wlinp : [O][HC] with col jj = dh*HP + j  <-  Wlin[o][dh*300 + j]
    for (int i = tid; i < O_ * HC; i += nth) {
        int o = i / HC, jj = i % HC;
        int dh = jj / HP, j = jj % HP;
        Wlinp[i] = (j < H_) ? Wlin[o * (2 * H_) + dh * H_ + j] : 0.f;
    }
}

// ---------------------------------------------------------------------------
// Embedding gather -> f16 [BT][EP], K-padded with zeros.
// ---------------------------------------------------------------------------
__global__ void embed_gather(const int* __restrict__ x, const float* __restrict__ table,
                             _Float16* __restrict__ emb)
{
    int m = blockIdx.x;          // 0..BT-1
    int e = threadIdx.x;         // 0..EP-1
    int idx = x[m];
    float v = (e < E_) ? table[(size_t)idx * E_ + e] : 0.f;
    emb[(size_t)m * EP + e] = (_Float16)v;
}

// ---------------------------------------------------------------------------
// Input projection: xp[t*B+b][n] = emb[b*T+t][:] . Wihp[n][:] + biasp[n]
// Each wave owns 1 m-tile x 4 n-tiles (16x64): A fragment reused 4x per
// k-step -> 4 WMMA per 10 b128 loads.  grid = (512, 19), block = 256 (8 waves).
// ---------------------------------------------------------------------------
__global__ void __launch_bounds__(256)
gemm_xp(const _Float16* __restrict__ A,   // [BT][EP]
        const _Float16* __restrict__ W,   // [NP][EP]
        const float*    __restrict__ bias,// [NP]
        _Float16*       __restrict__ xp)  // [T*B][NP]
{
    int wave = threadIdx.x >> 5;
    int lane = threadIdx.x & 31;
    int lm = lane & 15, hh = lane >> 4;
    int m0 = (blockIdx.x * 8 + wave) * 16;
    int n0 = blockIdx.y * 64;            // 4 n-tiles: n0 + 16*j

    v8f acc[4];
    #pragma unroll
    for (int j = 0; j < 4; ++j) {
        float bc = bias[n0 + 16 * j + lm];
        #pragma unroll
        for (int r = 0; r < 8; ++r) acc[j][r] = bc;
    }

    const _Float16* arow = A + (size_t)(m0 + lm) * EP;
    #pragma unroll
    for (int k0 = 0; k0 < EP; k0 += 32) {
        // A frag (16-bit A layout): lanes 0-15 K{0..7,16..23}, lanes 16-31 K{8..15,24..31}
        v8h alo = *(const v8h*)(arow + k0 + 8 * hh);
        v8h ahi = *(const v8h*)(arow + k0 + 16 + 8 * hh);
        v16h a;
        #pragma unroll
        for (int e = 0; e < 8; ++e) { a[e] = alo[e]; a[e + 8] = ahi[e]; }
        #pragma unroll
        for (int j = 0; j < 4; ++j) {
            // B frag: lanes 0-15 K{0..7,16..23}, lanes 16-31 K{8..15,24..31} (column n = W row n, K-major)
            const _Float16* brow = W + (size_t)(n0 + 16 * j + lm) * EP;
            v8h blo = *(const v8h*)(brow + k0 + 8 * hh);
            v8h bhi = *(const v8h*)(brow + k0 + 16 + 8 * hh);
            v16h b;
            #pragma unroll
            for (int e = 0; e < 8; ++e) { b[e] = blo[e]; b[e + 8] = bhi[e]; }
            acc[j] = __builtin_amdgcn_wmma_f32_16x16x32_f16(false, a, false, b, (short)0,
                                                            acc[j], false, false);
        }
    }
    #pragma unroll
    for (int j = 0; j < 4; ++j) {
        #pragma unroll
        for (int r = 0; r < 8; ++r) {
            int m  = m0 + r + 8 * hh;          // source row b*T + t
            int bb = m >> 9, t = m & (T_ - 1);
            xp[(size_t)(t * B_ + bb) * NP + n0 + 16 * j + lm] = (_Float16)acc[j][r];
        }
    }
}

// ---------------------------------------------------------------------------
// LSTM recurrence, one workgroup per direction (blockIdx.x = 0 fwd, 1 bwd).
// h lives in LDS as f16 [128][320]; c lives in VGPRs. Per step, each wave
// computes up to 5 column-groups; within a group the A fragment (rows of h)
// is loaded from LDS once per k-step and fed to 4 WMMAs (one per gate).
// Next step's xp tile is prefetched (global_prefetch_b8) behind the WMMAs.
// ---------------------------------------------------------------------------
__global__ void __launch_bounds__(1024)
lstm_dir(const _Float16* __restrict__ xp0,  // [T*B][NP] forward
         const _Float16* __restrict__ xp1,  // [T*B][NP] backward
         const _Float16* __restrict__ WhhAll, // [2][NP][KP]
         _Float16* __restrict__ hcat)       // [BT][HC]
{
    extern __shared__ _Float16 hsh[];       // [B_][KP] = 128*320 f16 = 80 KB
    const int d = blockIdx.x;
    const _Float16* xp  = d ? xp1 : xp0;
    const _Float16* Whh = WhhAll + (size_t)d * NP * KP;
    const int dirofs = d * HP;

    int tid = threadIdx.x;
    for (int i = tid; i < B_ * KP; i += blockDim.x) hsh[i] = (_Float16)0.f;
    __syncthreads();

    int wave = tid >> 5, lane = tid & 31;
    int lm = lane & 15, hh = lane >> 4;

    float cst[5][8];
    #pragma unroll
    for (int gi = 0; gi < 5; ++gi)
        #pragma unroll
        for (int r = 0; r < 8; ++r) cst[gi][r] = 0.f;

    for (int s = 0; s < T_; ++s) {
        int t = d ? (T_ - 1 - s) : s;
        _Float16 hnew[5][8];

        for (int gi = 0; gi < 5; ++gi) {
            int g = wave + 32 * gi;
            if (g >= NGROUPS) break;              // wave-uniform: EXEC stays full
            int m0 = (g & 7) * 16;                // batch tile
            int n0 = (g >> 3) * 16;               // hidden-column tile (within gate)

            // Prefetch next step's xp C-tile while this step computes.
            if (s + 1 < T_) {
                int tn = d ? (T_ - 2 - s) : (s + 1);
                const _Float16* pf = xp + (size_t)(tn * B_ + m0 + lm) * NP + n0;
                __builtin_prefetch(pf + (hh * 2 + 0) * HP, 0, 1);
                __builtin_prefetch(pf + (hh * 2 + 1) * HP, 0, 1);
            }

            // C init from xp (bias already folded in by gemm_xp)
            v8f acc[4];
            #pragma unroll
            for (int q = 0; q < 4; ++q)           // i, f, g, o
                #pragma unroll
                for (int r = 0; r < 8; ++r)
                    acc[q][r] = (float)xp[(size_t)(t * B_ + m0 + r + 8 * hh) * NP
                                          + q * HP + n0 + lm];

            // K loop: one LDS A-fragment feeds all 4 gate WMMAs.
            const _Float16* arow = hsh + (m0 + lm) * KP;
            #pragma unroll
            for (int k0 = 0; k0 < KP; k0 += 32) {
                v8h alo = *(const v8h*)(arow + k0 + 8 * hh);
                v8h ahi = *(const v8h*)(arow + k0 + 16 + 8 * hh);
                v16h a;
                #pragma unroll
                for (int e = 0; e < 8; ++e) { a[e] = alo[e]; a[e + 8] = ahi[e]; }
                #pragma unroll
                for (int q = 0; q < 4; ++q) {
                    const _Float16* brow = Whh + (size_t)(q * HP + n0 + lm) * KP;
                    v8h blo = *(const v8h*)(brow + k0 + 8 * hh);
                    v8h bhi = *(const v8h*)(brow + k0 + 16 + 8 * hh);
                    v16h b;
                    #pragma unroll
                    for (int e = 0; e < 8; ++e) { b[e] = blo[e]; b[e + 8] = bhi[e]; }
                    acc[q] = __builtin_amdgcn_wmma_f32_16x16x32_f16(false, a, false, b,
                                                                    (short)0, acc[q],
                                                                    false, false);
                }
            }

            #pragma unroll
            for (int r = 0; r < 8; ++r) {
                float iv = acc[0][r], fv = acc[1][r], gv = acc[2][r], ov = acc[3][r];
                float cc = sig_(fv) * cst[gi][r] + sig_(iv) * th_(gv);
                cst[gi][r] = cc;
                float hv = sig_(ov) * th_(cc);
                hnew[gi][r] = (_Float16)hv;
                int bb = m0 + r + 8 * hh;
                hcat[(size_t)(bb * T_ + t) * HC + dirofs + n0 + lm] = hnew[gi][r];
            }
        }
        __syncthreads();                           // all reads of old h done
        for (int gi = 0; gi < 5; ++gi) {
            int g = wave + 32 * gi;
            if (g >= NGROUPS) break;
            int m0 = (g & 7) * 16;
            int n0 = (g >> 3) * 16;
            #pragma unroll
            for (int r = 0; r < 8; ++r)
                hsh[(m0 + r + 8 * hh) * KP + n0 + lm] = hnew[gi][r];
        }
        __syncthreads();                           // new h visible to everyone
    }
}

// ---------------------------------------------------------------------------
// out[b,t,:] = softmax(hcat[b,t,:] @ Wlinp^T + blin).  O=9, K=608, mem-bound.
// ---------------------------------------------------------------------------
__global__ void __launch_bounds__(256)
linear_softmax(const _Float16* __restrict__ hcat,
               const float* __restrict__ Wlinp, const float* __restrict__ blin,
               float* __restrict__ out)
{
    __shared__ float wl[O_ * HC];
    __shared__ float bl[O_];
    for (int i = threadIdx.x; i < O_ * HC; i += blockDim.x) wl[i] = Wlinp[i];
    if (threadIdx.x < O_) bl[threadIdx.x] = blin[threadIdx.x];
    __syncthreads();

    int m = blockIdx.x * blockDim.x + threadIdx.x;
    if (m >= BT) return;
    float acc[O_];
    #pragma unroll
    for (int k = 0; k < O_; ++k) acc[k] = bl[k];
    const _Float16* hrow = hcat + (size_t)m * HC;
    for (int j = 0; j < HC; ++j) {
        float hv = (float)hrow[j];
        #pragma unroll
        for (int k = 0; k < O_; ++k) acc[k] += hv * wl[k * HC + j];
    }
    float mx = acc[0];
    #pragma unroll
    for (int k = 1; k < O_; ++k) mx = fmaxf(mx, acc[k]);
    float sum = 0.f;
    #pragma unroll
    for (int k = 0; k < O_; ++k) { acc[k] = __expf(acc[k] - mx); sum += acc[k]; }
    float inv = 1.f / sum;
    #pragma unroll
    for (int k = 0; k < O_; ++k) out[(size_t)m * O_ + k] = acc[k] * inv;
}

// ---------------------------------------------------------------------------
// Launch
// ---------------------------------------------------------------------------
extern "C" void kernel_launch(void* const* d_in, const int* in_sizes, int n_in,
                              void* d_out, int out_size, void* d_ws, size_t ws_size,
                              hipStream_t stream)
{
    const int*   x      = (const int*)  d_in[0];
    const float* table  = (const float*)d_in[1];
    const float* Wih_f  = (const float*)d_in[2];
    const float* Whh_f  = (const float*)d_in[3];
    const float* b_f    = (const float*)d_in[4];
    const float* Wih_b  = (const float*)d_in[5];
    const float* Whh_b  = (const float*)d_in[6];
    const float* b_b    = (const float*)d_in[7];
    const float* Wlin   = (const float*)d_in[8];
    const float* blin   = (const float*)d_in[9];

    char* ws = (char*)d_ws;
    size_t off = 0;
    auto alloc = [&](size_t bytes) -> void* {
        void* p = ws + off;
        off += (bytes + 255) & ~(size_t)255;
        return p;
    };
    _Float16* emb   = (_Float16*)alloc((size_t)BT * EP * sizeof(_Float16)); //  42 MB
    _Float16* Wihp  = (_Float16*)alloc((size_t)2 * NP * EP * sizeof(_Float16));
    _Float16* Whhp  = (_Float16*)alloc((size_t)2 * NP * KP * sizeof(_Float16));
    float*    biasp = (float*)   alloc((size_t)2 * NP * sizeof(float));
    float*    Wlinp = (float*)   alloc((size_t)O_ * HC * sizeof(float));
    _Float16* xp0   = (_Float16*)alloc((size_t)BT * NP * sizeof(_Float16)); // 159 MB
    _Float16* xp1   = (_Float16*)alloc((size_t)BT * NP * sizeof(_Float16)); // 159 MB
    _Float16* hcat  = (_Float16*)alloc((size_t)BT * HC * sizeof(_Float16)); //  76 MB

    prep_weights<<<256, 256, 0, stream>>>(Wih_f, Whh_f, b_f, Wih_b, Whh_b, b_b,
                                          Wlin, Wihp, Whhp, biasp, Wlinp);
    embed_gather<<<BT, EP, 0, stream>>>(x, table, emb);

    gemm_xp<<<dim3(512, 19), 256, 0, stream>>>(emb, Wihp,                   biasp,      xp0);
    gemm_xp<<<dim3(512, 19), 256, 0, stream>>>(emb, Wihp + (size_t)NP * EP, biasp + NP, xp1);

    lstm_dir<<<2, 1024, B_ * KP * sizeof(_Float16), stream>>>(xp0, xp1, Whhp, hcat);

    linear_softmax<<<(BT + 255) / 256, 256, 0, stream>>>(hcat, Wlinp, blin, (float*)d_out);
}
